// ChildSumTreeLSTM_29154238005598
// MI455X (gfx1250) — compile-verified, matrix operand order checked by reading, not run on hardware
//
#include <hip/hip_runtime.h>

typedef __attribute__((ext_vector_type(16))) __bf16        v16bf;
typedef __attribute__((ext_vector_type(8)))  float         v8f;
typedef __attribute__((ext_vector_type(4)))  unsigned int  u32x4;

#define TMEM   150
#define N3     450
#define NP     464     // 450 padded to 29*16
#define KX     320     // 300 padded to 10*32
#define KH     160     // 150 padded to 5*32
#define NLEAF  131072

// ---------------- helpers ----------------

__device__ __forceinline__ unsigned short f2bf(float f) {
  unsigned int u = __builtin_bit_cast(unsigned int, f);
  u += 0x7FFFu + ((u >> 16) & 1u);                    // round-to-nearest-even
  return (unsigned short)(u >> 16);
}
__device__ __forceinline__ float bf2f(unsigned short s) {
  unsigned int u = ((unsigned int)s) << 16;
  return __builtin_bit_cast(float, u);
}
__device__ __forceinline__ float sigf(float x)  { return 1.0f / (1.0f + __expf(-x)); }
__device__ __forceinline__ float tanhf_(float x){ return 1.0f - 2.0f / (__expf(2.0f * x) + 1.0f); }

// 16x32 bf16 WMMA A/B fragment from row-major [rows][stride] bf16 (global or LDS).
// Layout: lane l holds row (l&15); element e <-> K = 16*(e>>3) + 8*(l>>4) + (e&7)
// => two contiguous 16B loads per lane.
__device__ __forceinline__ v16bf load_frag(const unsigned short* __restrict__ base,
                                           int stride, int row, int k0, int lane) {
  const int r = row + (lane & 15);
  const int g = (lane >> 4) << 3;
  const unsigned short* p = base + (size_t)r * stride + k0 + g;
  union { u32x4 q[2]; v16bf v; } u;
  u.q[0] = *(const u32x4*)(p);
  u.q[1] = *(const u32x4*)(p + 16);
  return u.v;
}

#define WMMA_BF16(A, B, C) \
  __builtin_amdgcn_wmma_f32_16x16x32_bf16(false, (A), false, (B), (short)0, (C), false, false)

// ---------------- f32 -> padded bf16 conversion (weights only) ----------------

__global__ void convert_pad_bf16(const float* __restrict__ src, unsigned short* __restrict__ dst,
                                 int srows, int scols, int drows, int dcols) {
  const int total = drows * dcols;
  for (int idx = blockIdx.x * blockDim.x + threadIdx.x; idx < total;
       idx += gridDim.x * blockDim.x) {
    const int r = idx / dcols, c = idx - r * dcols;
    const float v = (r < srows && c < scols) ? src[(size_t)r * scols + c] : 0.0f;
    dst[(size_t)r * dcols + c] = f2bf(v);
  }
}

// ---------------- leaf level ----------------
// 32 leaf rows / WG. x converted f32->bf16 straight into LDS (no global bf16 copy).
// Each wave owns an N-column: B fragment loaded once, 2 WMMAs (2 M-tiles) per load.

__global__ void __launch_bounds__(256) leaf_kernel(
    const float* __restrict__ x,                  // [NLEAF][300] f32
    const unsigned short* __restrict__ wioux_b,   // [NP][KX] bf16
    const float* __restrict__ bioux, const float* __restrict__ biouh,
    float* __restrict__ c_out,                    // [NLEAF][TMEM] f32
    float* __restrict__ h_out,                    // d_out h_all base
    unsigned short* __restrict__ hch_out,         // [NLEAF][KH] bf16
    unsigned short* __restrict__ hsum_out) {      // [NLEAF/2][KH] bf16
  __shared__ __align__(16) unsigned short x_s[32][KX];   // 20 KB (overlaid by h_s later)
  __shared__ __align__(16) unsigned short iou_s[32][NP]; // 29 KB bf16
  const int tid  = threadIdx.x;
  const int wave = tid >> 5, lane = tid & 31;
  const int rowbase = blockIdx.x * 32;

  // stage x into LDS as zero-padded bf16
  for (int idx = tid; idx < 32 * KX; idx += 256) {
    const int r = idx / KX, c = idx - r * KX;
    const float v = (c < 300) ? x[(size_t)(rowbase + r) * 300 + c] : 0.0f;
    x_s[r][c] = f2bf(v);
  }
  __builtin_prefetch(wioux_b + (size_t)wave * 16 * KX, 0, 0);
  __syncthreads();

  for (int col = wave; col < 29; col += 8) {      // wave-uniform => EXEC all ones
    v8f a0 = {}, a1 = {};
    for (int k0 = 0; k0 < KX; k0 += 32) {
      const v16bf B  = load_frag(wioux_b, KX, col * 16, k0, lane);
      const v16bf A0 = load_frag(&x_s[0][0], KX, 0,  k0, lane);
      const v16bf A1 = load_frag(&x_s[0][0], KX, 16, k0, lane);
      a0 = WMMA_BF16(A0, B, a0);
      a1 = WMMA_BF16(A1, B, a1);
    }
    const int mr = (lane >> 4) * 8;
    const int nc = col * 16 + (lane & 15);
    for (int v = 0; v < 8; ++v) {
      iou_s[mr + v][nc]      = f2bf(a0[v]);
      iou_s[16 + mr + v][nc] = f2bf(a1[v]);
    }
  }
  __syncthreads();

  float (*h_s)[KH] = (float (*)[KH]) & x_s[0][0]; // 20 KB overlay (x no longer needed)

  for (int idx = tid; idx < 32 * KH; idx += 256) {
    const int p = idx / KH, j = idx - p * KH;
    const size_t row = (size_t)rowbase + p;
    if (j < TMEM) {
      const float vi = bf2f(iou_s[p][j])          + bioux[j]          + biouh[j];
      const float vo = bf2f(iou_s[p][j + TMEM])   + bioux[j + TMEM]   + biouh[j + TMEM];
      const float vu = bf2f(iou_s[p][j + 2*TMEM]) + bioux[j + 2*TMEM] + biouh[j + 2*TMEM];
      const float c = sigf(vi) * tanhf_(vu);      // leaf: fc term is exactly zero
      const float h = sigf(vo) * tanhf_(c);
      c_out[row * TMEM + j] = c;
      h_out[row * TMEM + j] = h;
      hch_out[row * KH + j] = f2bf(h);
      h_s[p][j] = h;
    } else {
      hch_out[row * KH + j] = 0;
      h_s[p][j] = 0.0f;
    }
  }
  __syncthreads();

  for (int idx = tid; idx < 16 * KH; idx += 256) { // child-sum h for next level
    const int q = idx / KH, j = idx - q * KH;
    const size_t prow = (size_t)blockIdx.x * 16 + q;
    hsum_out[prow * KH + j] = f2bf(h_s[2*q][j] + h_s[2*q + 1][j]);
  }
}

// ---------------- internal level ----------------
// 32 parents (64 children) / WG. 29 iou columns (2 M-tiles each) + 10 f columns
// (4 M-tiles each): every B fragment feeds 2 or 4 WMMAs.

__global__ void __launch_bounds__(256) level_kernel(
    int n,
    const unsigned short* __restrict__ hsum_in,   // [>=n][KH]  bf16
    const unsigned short* __restrict__ hch_in,    // [>=2n][KH] bf16
    const float* __restrict__ c_in,               // [2n][TMEM] f32
    const unsigned short* __restrict__ wiouh_b,   // [NP][KH] bf16
    const unsigned short* __restrict__ wfh_b,     // [KH][KH] bf16
    const float* __restrict__ biouh, const float* __restrict__ bfh,
    float* __restrict__ c_out, float* __restrict__ h_out,
    unsigned short* __restrict__ hch_out, unsigned short* __restrict__ hsum_out) {
  __shared__ __align__(16) unsigned short iou_s[32][NP]; // 29 KB bf16
  __shared__ __align__(16) unsigned short f_s[64][KH];   // 20 KB bf16
  __shared__ __align__(16) unsigned short h_s[32][KH];   // 10 KB bf16
  const int tid  = threadIdx.x;
  const int wave = tid >> 5, lane = tid & 31;
  const int pbase = blockIdx.x * 32;
  const int cbase = pbase * 2;

  __builtin_prefetch(wiouh_b + (size_t)wave * 16 * KH, 0, 0);

  for (int col = wave; col < 39; col += 8) {      // wave-uniform branch
    if (col < 29) {                               // iou = hsum @ Wiouh.T
      v8f a0 = {}, a1 = {};
      for (int k0 = 0; k0 < KH; k0 += 32) {
        const v16bf B  = load_frag(wiouh_b, KH, col * 16, k0, lane);
        const v16bf A0 = load_frag(hsum_in, KH, pbase,      k0, lane);
        const v16bf A1 = load_frag(hsum_in, KH, pbase + 16, k0, lane);
        a0 = WMMA_BF16(A0, B, a0);
        a1 = WMMA_BF16(A1, B, a1);
      }
      const int mr = (lane >> 4) * 8;
      const int nc = col * 16 + (lane & 15);
      for (int v = 0; v < 8; ++v) {
        iou_s[mr + v][nc]      = f2bf(a0[v]);
        iou_s[16 + mr + v][nc] = f2bf(a1[v]);
      }
    } else {                                      // f = child_h @ Wfh.T
      const int nt = col - 29;
      v8f a0 = {}, a1 = {}, a2 = {}, a3 = {};
      for (int k0 = 0; k0 < KH; k0 += 32) {
        const v16bf B  = load_frag(wfh_b, KH, nt * 16, k0, lane);
        const v16bf A0 = load_frag(hch_in, KH, cbase,      k0, lane);
        const v16bf A1 = load_frag(hch_in, KH, cbase + 16, k0, lane);
        const v16bf A2 = load_frag(hch_in, KH, cbase + 32, k0, lane);
        const v16bf A3 = load_frag(hch_in, KH, cbase + 48, k0, lane);
        a0 = WMMA_BF16(A0, B, a0);
        a1 = WMMA_BF16(A1, B, a1);
        a2 = WMMA_BF16(A2, B, a2);
        a3 = WMMA_BF16(A3, B, a3);
      }
      const int mr = (lane >> 4) * 8;
      const int nc = nt * 16 + (lane & 15);
      for (int v = 0; v < 8; ++v) {
        f_s[mr + v][nc]      = f2bf(a0[v]);
        f_s[16 + mr + v][nc] = f2bf(a1[v]);
        f_s[32 + mr + v][nc] = f2bf(a2[v]);
        f_s[48 + mr + v][nc] = f2bf(a3[v]);
      }
    }
  }
  __syncthreads();

  for (int idx = tid; idx < 32 * KH; idx += 256) {
    const int p = idx / KH, j = idx - p * KH;
    const size_t row = (size_t)pbase + p;
    float h = 0.0f;
    if ((int)row < n && j < TMEM) {
      const float vi = bf2f(iou_s[p][j])          + biouh[j];
      const float vo = bf2f(iou_s[p][j + TMEM])   + biouh[j + TMEM];
      const float vu = bf2f(iou_s[p][j + 2*TMEM]) + biouh[j + 2*TMEM];
      const float f0 = sigf(bf2f(f_s[2*p][j])     + bfh[j]);
      const float f1 = sigf(bf2f(f_s[2*p + 1][j]) + bfh[j]);
      const float c = sigf(vi) * tanhf_(vu)
                    + f0 * c_in[(row * 2)     * TMEM + j]
                    + f1 * c_in[(row * 2 + 1) * TMEM + j];
      h = sigf(vo) * tanhf_(c);
      c_out[row * TMEM + j] = c;
      h_out[row * TMEM + j] = h;
    }
    if ((int)row < n) hch_out[row * KH + j] = (j < TMEM) ? f2bf(h) : (unsigned short)0;
    h_s[p][j] = f2bf(h);
  }
  __syncthreads();

  for (int idx = tid; idx < 16 * KH; idx += 256) {
    const int q = idx / KH, j = idx - q * KH;
    const size_t prow = (size_t)blockIdx.x * 16 + q;
    hsum_out[prow * KH + j] = f2bf(bf2f(h_s[2*q][j]) + bf2f(h_s[2*q + 1][j]));
  }
}

// ---------------- root pass-through ----------------

__global__ void root_copy(const float* __restrict__ c_last, float* __restrict__ out) {
  const int idx = blockIdx.x * blockDim.x + threadIdx.x;
  if (idx < 300) {
    out[idx]       = c_last[idx];                            // root_c = children c
    out[300 + idx] = out[600 + (size_t)262140 * TMEM + idx]; // root_h = last level h
  }
}

// ---------------- launcher ----------------

extern "C" void kernel_launch(void* const* d_in, const int* in_sizes, int n_in,
                              void* d_out, int out_size, void* d_ws, size_t ws_size,
                              hipStream_t stream) {
  const float* x     = (const float*)d_in[0];
  const float* Wioux = (const float*)d_in[1];
  const float* bioux = (const float*)d_in[2];
  const float* Wiouh = (const float*)d_in[3];
  const float* biouh = (const float*)d_in[4];
  const float* Wfh   = (const float*)d_in[7];   // Wfx/bfx unused (fc==0 at leaves)
  const float* bfh   = (const float*)d_in[8];
  float* out = (float*)d_out;

  char* wsp = (char*)d_ws;
  auto alloc = [&](size_t bytes) -> char* {
    char* p = wsp; wsp += (bytes + 255) & ~(size_t)255; return p;
  };
  unsigned short* wioux_b = (unsigned short*)alloc((size_t)NP * KX * 2);
  unsigned short* wiouh_b = (unsigned short*)alloc((size_t)NP * KH * 2);
  unsigned short* wfh_b   = (unsigned short*)alloc((size_t)KH * KH * 2);
  unsigned short* hch0    = (unsigned short*)alloc((size_t)NLEAF * KH * 2);
  unsigned short* hch1    = (unsigned short*)alloc((size_t)(NLEAF / 2) * KH * 2);
  unsigned short* hs0     = (unsigned short*)alloc((size_t)(NLEAF / 2) * KH * 2);
  unsigned short* hs1     = (unsigned short*)alloc((size_t)(NLEAF / 4) * KH * 2);
  float*          c0      = (float*)alloc((size_t)NLEAF * TMEM * 4);
  float*          c1      = (float*)alloc((size_t)(NLEAF / 2) * TMEM * 4);

  // weight bf16 conversion + zero padding (x conversion is fused into leaf_kernel)
  convert_pad_bf16<<<64, 256, 0, stream>>>(Wioux, wioux_b, N3,   300,  NP, KX);
  convert_pad_bf16<<<32, 256, 0, stream>>>(Wiouh, wiouh_b, N3,   TMEM, NP, KH);
  convert_pad_bf16<<<16, 256, 0, stream>>>(Wfh,   wfh_b,   TMEM, TMEM, KH, KH);

  float* h_all = out + 600;
  leaf_kernel<<<NLEAF / 32, 256, 0, stream>>>(x, wioux_b, bioux, biouh,
                                              c0, h_all, hch0, hs0);

  unsigned short* hch[2] = {hch0, hch1};
  unsigned short* hs[2]  = {hs0, hs1};
  float*          cb[2]  = {c0, c1};
  size_t hoff = NLEAF;
  int cur = 0;
  for (int n = NLEAF / 2; n >= 2; n >>= 1) {    // 16 internal levels
    const int nxt = cur ^ 1;
    level_kernel<<<(n + 31) / 32, 256, 0, stream>>>(
        n, hs[cur], hch[cur], cb[cur], wiouh_b, wfh_b, biouh, bfh,
        cb[nxt], h_all + hoff * TMEM, hch[nxt], hs[nxt]);
    hoff += n;
    cur = nxt;
  }

  root_copy<<<2, 256, 0, stream>>>(cb[cur], out);
}
